// GAAE_mod1_66657892434572
// MI455X (gfx1250) — compile-verified, matrix operand order checked by reading, not run on hardware
//
#include <hip/hip_runtime.h>
#include <math.h>

// ---------------- problem constants (match reference) ----------------
static constexpr int NN      = 30000;
static constexpr int DEG     = 10;
static constexpr int IN_DIM  = 3000;
static constexpr int DEG_DIM = 512;
static constexpr int HID     = 512;
static constexpr int OUT_DIM = 30;
static constexpr int PRED    = 20;
static constexpr float NEG_SLOPE = 0.2f;

// padded dims
static constexpr int MP     = 30080;  // 235 * 128
static constexpr int KP_IN  = 3008;   // 47 * 64  (3000 padded)
static constexpr int KP_SM  = 64;     // 30/20 padded
static constexpr int NP_SM  = 128;    // 30/20 padded to one BN tile
static constexpr int NP_DEC = 3072;   // 3000 padded

// ---------------- CDNA5 async-copy availability ----------------
#if defined(__has_builtin)
#  if __has_builtin(__builtin_amdgcn_global_load_async_to_lds_b128) && \
      __has_builtin(__builtin_amdgcn_s_wait_asynccnt)
#    define USE_ASYNC_LDS 1
#  endif
#endif
#ifndef USE_ASYNC_LDS
#  define USE_ASYNC_LDS 0
#endif

// ---------------- types ----------------
typedef __attribute__((ext_vector_type(16))) __bf16          v16bf;
typedef __attribute__((ext_vector_type(8)))  float           v8f;
typedef __attribute__((ext_vector_type(8)))  unsigned short  ushort8v;
typedef __attribute__((ext_vector_type(4)))  unsigned short  ushort4v;
typedef __attribute__((ext_vector_type(4)))  int             int4v;
#if USE_ASYNC_LDS
typedef __attribute__((address_space(3))) int4v* lds_v4p;
#endif

union Frag { v16bf bf; ushort8v u[2]; };

__device__ __forceinline__ unsigned short f2bf_rne(float f) {
    union { float f; unsigned u; } x; x.f = f;
    unsigned r = x.u + 0x7FFFu + ((x.u >> 16) & 1u);
    return (unsigned short)(r >> 16);
}
__device__ __forceinline__ float bfbits2f(unsigned hi16bits) {
    union { unsigned u; float f; } x; x.u = hi16bits; return x.f;
}
__device__ __forceinline__ float elu1(float v) {
    return v > 0.f ? v : (__expf(v) - 1.f);
}

// ---------------- pad/transpose/convert f32 -> bf16 ----------------
__global__ __launch_bounds__(256)
void convert_pad_bf16(const float* __restrict__ src, unsigned short* __restrict__ dst,
                      int NrowsP, int Kp, int Ntrue, int Ktrue, int lds, int trans)
{
    int idx = blockIdx.x * blockDim.x + threadIdx.x;       // unit of 4 halves
    int units = (NrowsP * Kp) >> 2;
    if (idx >= units) return;
    int kq = (idx % (Kp >> 2)) << 2;
    int n  = idx / (Kp >> 2);
    ushort4v h;
    #pragma unroll
    for (int j = 0; j < 4; ++j) {
        int k = kq + j;
        float v = 0.f;
        if (n < Ntrue && k < Ktrue)
            v = trans ? src[(size_t)n * lds + k] : src[(size_t)k * lds + n];
        h[j] = f2bf_rne(v);
    }
    *(ushort4v*)&dst[(size_t)n * Kp + kq] = h;
}

// ---------------- bf16 WMMA GEMM on padded operands ----------------
// A: bf16 [MP][Kp], Bt: bf16 [Np][Kp] (pre-transposed), Kp%64==0, Np%128==0.
// outBf16==1: C = bf16 [MP][Np] (unguarded); outBf16==0: C = f32 [M][N] true layout.
static constexpr int BM = 128, BN = 128, BK = 64, PADK = 72;  // 144B row stride

#if USE_ASYNC_LDS
static constexpr int NBUF = 2;
#else
static constexpr int NBUF = 1;
#endif

__global__ __launch_bounds__(256)
void gemm_bf16p(const unsigned short* __restrict__ A, const unsigned short* __restrict__ Bt,
                const float* __restrict__ bias, void* __restrict__ Cout,
                int M, int N, int Kp, int Np, int outBf16, int act)
{
    __shared__ unsigned short As[NBUF][BM * PADK];
    __shared__ unsigned short Bs[NBUF][BN * PADK];

    const int tid   = threadIdx.x;
    const int lane  = tid & 31;
    const int wave  = tid >> 5;
    const int waveM = wave & 3;       // 4 row-waves of 32 rows
    const int waveN = wave >> 2;      // 2 col-waves of 64 cols
    const int m0 = blockIdx.y * BM;
    const int n0 = blockIdx.x * BN;

    const int l16  = lane & 15;
    const int half = lane >> 4;

    // per-thread staging coordinates (4 b128 units per matrix)
    int sr[4], sc[4];
    #pragma unroll
    for (int i = 0; i < 4; ++i) {
        int u = tid + i * 256;
        sr[i] = u >> 3;
        sc[i] = (u & 7) << 3;
    }

    v8f acc[2][4] = {};

    auto computeTile = [&](const unsigned short* Ab, const unsigned short* Bb) {
        #pragma unroll
        for (int s = 0; s < 2; ++s) {
            const int kb0 = s * 32 + 8 * half;
            const int kb1 = s * 32 + 16 + 8 * half;
            Frag afr[2], bfr[4];
            #pragma unroll
            for (int mt = 0; mt < 2; ++mt) {
                int row = waveM * 32 + mt * 16 + l16;
                afr[mt].u[0] = *(const ushort8v*)(&Ab[row * PADK + kb0]);
                afr[mt].u[1] = *(const ushort8v*)(&Ab[row * PADK + kb1]);
            }
            #pragma unroll
            for (int nt = 0; nt < 4; ++nt) {
                int col = waveN * 64 + nt * 16 + l16;
                bfr[nt].u[0] = *(const ushort8v*)(&Bb[col * PADK + kb0]);
                bfr[nt].u[1] = *(const ushort8v*)(&Bb[col * PADK + kb1]);
            }
            #pragma unroll
            for (int mt = 0; mt < 2; ++mt)
                #pragma unroll
                for (int nt = 0; nt < 4; ++nt)
                    acc[mt][nt] = __builtin_amdgcn_wmma_f32_16x16x32_bf16(
                        false, afr[mt].bf, false, bfr[nt].bf,
                        (short)0, acc[mt][nt], false, false);
        }
    };

    const int ksteps = Kp >> 6;

#if USE_ASYNC_LDS
    // ---- async double-buffered pipeline: ASYNCcnt-tracked LDS DMA ----
    auto issue = [&](int k0, int b) {
        #pragma unroll
        for (int i = 0; i < 4; ++i) {
            __builtin_amdgcn_global_load_async_to_lds_b128(
                (int4v*)&A[(size_t)(m0 + sr[i]) * Kp + (k0 + sc[i])],
                (lds_v4p)(unsigned)(size_t)&As[b][sr[i] * PADK + sc[i]],
                0, 0);
            __builtin_amdgcn_global_load_async_to_lds_b128(
                (int4v*)&Bt[(size_t)(n0 + sr[i]) * Kp + (k0 + sc[i])],
                (lds_v4p)(unsigned)(size_t)&Bs[b][sr[i] * PADK + sc[i]],
                0, 0);
        }
    };
    issue(0, 0);
    for (int ks = 0; ks < ksteps; ++ks) {
        const int bsel = ks & 1;
        if (ks + 1 < ksteps) {
            issue((ks + 1) << 6, bsel ^ 1);              // 8 asyncs for next tile
            __builtin_amdgcn_s_wait_asynccnt(8);         // own tile ks complete
        } else {
            __builtin_amdgcn_s_wait_asynccnt(0);
        }
        __syncthreads();                                 // tile ks visible to all waves
        computeTile(&As[bsel][0], &Bs[bsel][0]);
        __syncthreads();                                 // done reading buffer bsel
    }
#else
    // ---- register double-buffered pipeline ----
    ushort8v ta[4], tb[4];
    auto gload = [&](int k0) {
        #pragma unroll
        for (int i = 0; i < 4; ++i) {
            ta[i] = *(const ushort8v*)&A[(size_t)(m0 + sr[i]) * Kp + (k0 + sc[i])];
            tb[i] = *(const ushort8v*)&Bt[(size_t)(n0 + sr[i]) * Kp + (k0 + sc[i])];
        }
    };
    auto lstore = [&]() {
        #pragma unroll
        for (int i = 0; i < 4; ++i) {
            *(ushort8v*)&As[0][sr[i] * PADK + sc[i]] = ta[i];
            *(ushort8v*)&Bs[0][sr[i] * PADK + sc[i]] = tb[i];
        }
    };
    gload(0);
    lstore();
    __syncthreads();
    for (int ks = 0; ks < ksteps; ++ks) {
        if (ks + 1 < ksteps) gload((ks + 1) << 6);       // in flight during compute
        computeTile(&As[0][0], &Bs[0][0]);
        __syncthreads();
        if (ks + 1 < ksteps) lstore();
        __syncthreads();
    }
#endif

    // ---- epilogue: n = lane%16, m = r + 8*half ----
    if (outBf16) {
        unsigned short* C = (unsigned short*)Cout;       // [MP][Np]
        #pragma unroll
        for (int nt = 0; nt < 4; ++nt) {
            int gn = n0 + waveN * 64 + nt * 16 + l16;
            float bv = bias ? bias[gn] : 0.f;
            #pragma unroll
            for (int mt = 0; mt < 2; ++mt)
                #pragma unroll
                for (int r = 0; r < 8; ++r) {
                    int gm = m0 + waveM * 32 + mt * 16 + r + 8 * half;
                    float v = acc[mt][nt][r] + bv;
                    if (act == 1) v = elu1(v);
                    C[(size_t)gm * Np + gn] = f2bf_rne(v);
                }
        }
    } else {
        float* C = (float*)Cout;                         // [M][N] true layout
        const bool mFull = (m0 + BM <= M);
        const bool nFull = (n0 + BN <= N);
        if (mFull && nFull) {
            #pragma unroll
            for (int nt = 0; nt < 4; ++nt) {
                int gn = n0 + waveN * 64 + nt * 16 + l16;
                float bv = bias ? bias[gn] : 0.f;
                #pragma unroll
                for (int mt = 0; mt < 2; ++mt)
                    #pragma unroll
                    for (int r = 0; r < 8; ++r) {
                        int gm = m0 + waveM * 32 + mt * 16 + r + 8 * half;
                        float v = acc[mt][nt][r] + bv;
                        if (act == 1) v = elu1(v);
                        C[(size_t)gm * N + gn] = v;
                    }
            }
        } else {
            #pragma unroll
            for (int nt = 0; nt < 4; ++nt) {
                int gn = n0 + waveN * 64 + nt * 16 + l16;
                if (gn >= N) continue;
                float bv = bias ? bias[gn] : 0.f;
                #pragma unroll
                for (int mt = 0; mt < 2; ++mt)
                    #pragma unroll
                    for (int r = 0; r < 8; ++r) {
                        int gm = m0 + waveM * 32 + mt * 16 + r + 8 * half;
                        if (gm < M) {
                            float v = acc[mt][nt][r] + bv;
                            if (act == 1) v = elu1(v);
                            C[(size_t)gm * N + gn] = v;
                        }
                    }
            }
        }
    }
}

// ---------------- per-row attention logits (bf16 x) ----------------
__global__ __launch_bounds__(256)
void row_logits(const unsigned short* __restrict__ x, const float* __restrict__ avs,
                const float* __restrict__ avd, float* __restrict__ outs,
                float* __restrict__ outd, int n, int hid)
{
    int wave = threadIdx.x >> 5;
    int lane = threadIdx.x & 31;
    int row  = blockIdx.x * 8 + wave;
    if (row >= n) return;
    const unsigned short* xr = x + (size_t)row * hid;
    float s = 0.f, d = 0.f;
    for (int k = lane * 2; k < hid; k += 64) {
        unsigned v2 = *(const unsigned*)&xr[k];
        float v0 = bfbits2f(v2 << 16);
        float v1 = bfbits2f(v2 & 0xFFFF0000u);
        s += v0 * avs[k] + v1 * avs[k + 1];
        d += v0 * avd[k] + v1 * avd[k + 1];
    }
    #pragma unroll
    for (int off = 16; off > 0; off >>= 1) {
        s += __shfl_xor(s, off, 32);
        d += __shfl_xor(d, off, 32);
    }
    if (lane == 0) { outs[row] = s; outd[row] = d; }
}

// ---------------- segment-softmax attention aggregate + ELU (bf16 in/out) ----------------
__global__ __launch_bounds__(256)
void attn_agg_elu(const unsigned short* __restrict__ x, const float* __restrict__ as_,
                  const float* __restrict__ ad_, const int* __restrict__ srcs,
                  unsigned short* __restrict__ out, int n, int hid)
{
    __shared__ int   s_src[DEG];
    __shared__ float s_e[DEG];
    __shared__ float s_alpha[DEG];

    const int node = blockIdx.x;
    const int tid  = threadIdx.x;

    if (tid < DEG) {
        int s = srcs[node * DEG + tid];
        s_src[tid] = s;
        float e = as_[s] + ad_[node];
        s_e[tid] = e > 0.f ? e : NEG_SLOPE * e;   // leaky_relu
    }
    __syncthreads();
    if (tid < DEG) {
        float m = -3.4e38f;
        #pragma unroll
        for (int j = 0; j < DEG; ++j) m = fmaxf(m, s_e[j]);
        float denom = 0.f;
        #pragma unroll
        for (int j = 0; j < DEG; ++j) denom += __expf(s_e[j] - m);
        s_alpha[tid] = __expf(s_e[tid] - m) / denom;  // denom >= 1
    }
    __syncthreads();

    for (int c2 = tid * 2; c2 < hid; c2 += blockDim.x * 2) {
        float a0 = 0.f, a1 = 0.f;
        #pragma unroll
        for (int j = 0; j < DEG; ++j) {
            unsigned v2 = *(const unsigned*)&x[(size_t)s_src[j] * hid + c2];
            float al = s_alpha[j];
            a0 += al * bfbits2f(v2 << 16);
            a1 += al * bfbits2f(v2 & 0xFFFF0000u);
        }
        unsigned o = (unsigned)f2bf_rne(elu1(a0)) | ((unsigned)f2bf_rne(elu1(a1)) << 16);
        *(unsigned*)&out[(size_t)node * hid + c2] = o;
    }
}

// ---------------- log_softmax over PRED=20 ----------------
__global__ __launch_bounds__(256)
void log_softmax_rows(const float* __restrict__ in, float* __restrict__ out, int n)
{
    int row = blockIdx.x * blockDim.x + threadIdx.x;
    if (row >= n) return;
    float v[PRED];
    const float* r = in + (size_t)row * PRED;
    float m = -3.4e38f;
    #pragma unroll
    for (int i = 0; i < PRED; ++i) { v[i] = r[i]; m = fmaxf(m, v[i]); }
    float s = 0.f;
    #pragma unroll
    for (int i = 0; i < PRED; ++i) s += __expf(v[i] - m);
    float ls = __logf(s) + m;
    float* o = out + (size_t)row * PRED;
    #pragma unroll
    for (int i = 0; i < PRED; ++i) o[i] = v[i] - ls;
}

// ---------------- host launcher ----------------
extern "C" void kernel_launch(void* const* d_in, const int* in_sizes, int n_in,
                              void* d_out, int out_size, void* d_ws, size_t ws_size,
                              hipStream_t stream) {
    const float* features = (const float*)d_in[0];
    const int*   src      = (const int*)d_in[1];   // edge_index row0 = src
    const float* W_enc    = (const float*)d_in[2];
    const float* b_enc    = (const float*)d_in[3];
    const float* W1       = (const float*)d_in[4];
    const float* att_s    = (const float*)d_in[5];
    const float* att_d    = (const float*)d_in[6];
    const float* W2       = (const float*)d_in[7];
    const float* W_pred   = (const float*)d_in[8];
    const float* b_pred   = (const float*)d_in[9];
    const float* W_dec    = (const float*)d_in[10];
    const float* b_dec    = (const float*)d_in[11];
    (void)in_sizes; (void)n_in; (void)out_size; (void)ws_size;

    // ---- workspace carve-up (256B aligned chunks) ----
    char* w = (char*)d_ws;
    auto alloc = [&](size_t bytes) { void* p = (void*)w; w += (bytes + 255) & ~(size_t)255; return p; };
    unsigned short* feat_bf = (unsigned short*)alloc((size_t)MP * KP_IN * 2);
    unsigned short* bufA    = (unsigned short*)alloc((size_t)MP * HID * 2);  // de -> x3
    unsigned short* bufB    = (unsigned short*)alloc((size_t)MP * HID * 2);  // x1 -> h3
    unsigned short* bufC    = (unsigned short*)alloc((size_t)MP * HID * 2);  // h1 -> elu(h4)
    unsigned short* h2bf    = (unsigned short*)alloc((size_t)MP * KP_SM * 2);
    unsigned short* wEnc_t  = (unsigned short*)alloc((size_t)DEG_DIM * KP_IN * 2);
    unsigned short* w1_t    = (unsigned short*)alloc((size_t)HID * HID * 2);
    unsigned short* w1_d    = (unsigned short*)alloc((size_t)HID * HID * 2);
    unsigned short* w2_t    = (unsigned short*)alloc((size_t)NP_SM * HID * 2);
    unsigned short* w2_d    = (unsigned short*)alloc((size_t)HID * KP_SM * 2);
    unsigned short* wPred_t = (unsigned short*)alloc((size_t)NP_SM * KP_SM * 2);
    unsigned short* wDec_t  = (unsigned short*)alloc((size_t)NP_DEC * HID * 2);
    float* predb = (float*)alloc((size_t)NN * PRED * 4);
    float* a1s   = (float*)alloc((size_t)NN * 4);
    float* a1d   = (float*)alloc((size_t)NN * 4);

    // output layout: h2 [NN,30] | out [NN,3000] | log_softmax(pred) [NN,20]
    float* h2     = (float*)d_out;
    float* outbuf = h2 + (size_t)NN * OUT_DIM;
    float* lsm    = outbuf + (size_t)NN * IN_DIM;

    auto conv = [&](const float* s, unsigned short* dd, int NrowsP, int Kp,
                    int Ntrue, int Ktrue, int lds, int trans) {
        int units = (NrowsP * Kp) >> 2;
        convert_pad_bf16<<<(units + 255) / 256, 256, 0, stream>>>(s, dd, NrowsP, Kp, Ntrue, Ktrue, lds, trans);
    };
    auto gemm = [&](const unsigned short* A, const unsigned short* Bt, const float* bias,
                    void* C, int M, int Nc, int Kp, int Np, int outBf16, int act) {
        dim3 g(Np / BN, MP / BM);
        gemm_bf16p<<<g, 256, 0, stream>>>(A, Bt, bias, C, M, Nc, Kp, Np, outBf16, act);
    };

    // ---- operand conversion ----
    conv(features, feat_bf, MP, KP_IN, NN, IN_DIM, IN_DIM, 1);
    conv(W_enc,  wEnc_t,  DEG_DIM, KP_IN, DEG_DIM, IN_DIM, DEG_DIM, 0);
    conv(W1,     w1_t,    HID, HID, HID, HID, HID, 0);
    conv(W1,     w1_d,    HID, HID, HID, HID, HID, 1);
    conv(W2,     w2_t,    NP_SM, HID, OUT_DIM, HID, OUT_DIM, 0);
    conv(W2,     w2_d,    HID, KP_SM, HID, OUT_DIM, OUT_DIM, 1);
    conv(W_pred, wPred_t, NP_SM, KP_SM, PRED, OUT_DIM, PRED, 0);
    conv(W_dec,  wDec_t,  NP_DEC, HID, IN_DIM, HID, IN_DIM, 0);

    // 1) de = features @ W_enc + b_enc        -> bufA bf16 [MP][512]
    gemm(feat_bf, wEnc_t, b_enc, bufA, NN, DEG_DIM, KP_IN, DEG_DIM, 1, 0);
    // 2) x1 = de @ W1                          -> bufB bf16
    gemm(bufA, w1_t, nullptr, bufB, NN, HID, HID, HID, 1, 0);
    // 3) a1s/a1d
    row_logits<<<(NN + 7) / 8, 256, 0, stream>>>(bufB, att_s, att_d, a1s, a1d, NN, HID);
    // 4) h1 = elu(attn_aggregate(x1))          -> bufC bf16
    attn_agg_elu<<<NN, 256, 0, stream>>>(bufB, a1s, a1d, src, bufC, NN, HID);
    // 5) h2 = h1 @ W2                          -> d_out f32 [NN,30]
    gemm(bufC, w2_t, nullptr, h2, NN, OUT_DIM, HID, NP_SM, 0, 0);
    // 6) h2 -> padded bf16 [MP][64]
    conv(h2, h2bf, MP, KP_SM, NN, OUT_DIM, OUT_DIM, 1);
    // 7) pred = h2 @ W_pred + b_pred           -> predb f32 [NN,20]
    gemm(h2bf, wPred_t, b_pred, predb, NN, PRED, KP_SM, NP_SM, 0, 0);
    // 8) log_softmax(pred)                     -> d_out tail
    log_softmax_rows<<<(NN + 255) / 256, 256, 0, stream>>>(predb, lsm, NN);
    // 9) x3 = h2 @ W2^T                        -> bufA bf16
    gemm(h2bf, w2_d, nullptr, bufA, NN, HID, KP_SM, HID, 1, 0);
    // 10) h3 = elu(attn_aggregate(x3))         -> bufB bf16
    attn_agg_elu<<<NN, 256, 0, stream>>>(bufA, a1s, a1d, src, bufB, NN, HID);
    // 11) elu(h4) = elu(h3 @ W1^T)             -> bufC bf16 (ELU fused)
    gemm(bufB, w1_d, nullptr, bufC, NN, DEG_DIM, HID, HID, 1, 1);
    // 12) out = elu(h4) @ W_dec + b_dec        -> d_out f32 [NN,3000]
    gemm(bufC, wDec_t, b_dec, outbuf, NN, IN_DIM, HID, NP_DEC, 0, 0);
}
